// MambaBlock_61435212202224
// MI455X (gfx1250) — compile-verified, matrix operand order checked by reading, not run on hardware
//
#include <hip/hip_runtime.h>
#include <hip/hip_bf16.h>

// ---------------------------------------------------------------------------
// Mamba block for MI455X (gfx1250): wave32, bf16 WMMA matrix pipe, async
// global->LDS tile staging (ASYNCcnt), double-buffered LDS, peeled prologue/
// epilogue so the steady-state loop is branch-free.
//   P0: x -> Xb (bf16);  W_in/W_dt/W_out -> transposed bf16 [N][K]
//   K1: xz = x @ W_in + b_in ; silu -> XC fp32 + XCb bf16 | ZS fp32
//   K2: BC = XC @ W_xproj + b_xproj           (tiny N=32, VALU)
//   K3: DT = softplus(XCb @ W_dtT + b_dt)     (WMMA)
//   K4: selective scan -> Y fp32 (aliases DT) + Yb bf16
//   K5: out = Yb @ W_outT + b_out             (WMMA)
// ---------------------------------------------------------------------------

typedef __attribute__((ext_vector_type(4)))  float  v4f;
typedef __attribute__((ext_vector_type(8)))  float  v8f;
typedef __attribute__((ext_vector_type(4)))  __bf16 v4bf;
typedef __attribute__((ext_vector_type(8)))  __bf16 v8bf;
typedef __attribute__((ext_vector_type(16))) __bf16 v16bf;

#define D_MODEL 1024
#define D_INNER 2048
#define D_STATE 16
#define BATCH   4
#define SEQ     2048
#define MTOK    (BATCH * SEQ)   // 8192 token rows

__device__ __forceinline__ __bf16 f2bf(float f) {
  union { float f; unsigned u; } c; c.f = f;
  unsigned r = c.u + 0x7FFFu + ((c.u >> 16) & 1u);   // round-to-nearest-even
  unsigned short h = (unsigned short)(r >> 16);
  __bf16 o; __builtin_memcpy(&o, &h, 2);
  return o;
}
__device__ __forceinline__ float sigmoidf_(float x) { return 1.f / (1.f + __expf(-x)); }
__device__ __forceinline__ float siluf_(float x)    { return x * sigmoidf_(x); }
__device__ __forceinline__ float softplusf_(float x){ return fmaxf(x, 0.f) + log1pf(__expf(-fabsf(x))); }

// LDS byte offset of a generic pointer into __shared__ (addrspace(3) ptrtoint).
typedef __attribute__((address_space(3))) const unsigned char lds_cbyte;
__device__ __forceinline__ unsigned lds_addr(const void* p) {
  return (unsigned)(unsigned long long)(lds_cbyte*)p;
}

// 16B async copy global -> LDS (GLOBAL_LOAD_ASYNC_TO_LDS_B128, ASYNCcnt).
__device__ __forceinline__ void async_copy_b128(unsigned lds_off, const void* gptr) {
  asm volatile("global_load_async_to_lds_b128 %0, %1, off"
               :: "v"(lds_off), "v"(gptr) : "memory");
}
__device__ __forceinline__ void wait_async0() {
  asm volatile("s_wait_asynccnt 0x0" ::: "memory");
}

// Fragment gather from LDS row (stored along K): two 16B ds_load_b128.
// Element e of the v16bf maps to K = (e>>3)*16 + khalf*8 + (e&7).
__device__ __forceinline__ v16bf load_frag(const __bf16* rowp, int khalf) {
  v8bf lo = *(const v8bf*)(rowp + khalf * 8);
  v8bf hi = *(const v8bf*)(rowp + 16 + khalf * 8);
  return __builtin_shufflevector(lo, hi, 0,1,2,3,4,5,6,7,8,9,10,11,12,13,14,15);
}

// ---------------------------------------------------------------------------
// Prep kernels (one-time per launch, negligible vs. 172 GFLOP of GEMMs)
// ---------------------------------------------------------------------------
__global__ __launch_bounds__(256)
void cvt_f32_bf16(const float* __restrict__ src, __bf16* __restrict__ dst, int n4) {
  int i = blockIdx.x * blockDim.x + threadIdx.x;
  if (i >= n4) return;
  v4f v = *(const v4f*)(src + (size_t)i * 4);
  v4bf o = { f2bf(v[0]), f2bf(v[1]), f2bf(v[2]), f2bf(v[3]) };
  *(v4bf*)(dst + (size_t)i * 4) = o;
}

// dst[c*R + r] = bf16(src[r*C + c])  : weight [K][N] -> [N][K] bf16
__global__ __launch_bounds__(256)
void transpose_cvt(const float* __restrict__ src, __bf16* __restrict__ dst,
                   int R, int C) {
  __shared__ float tile[32][33];
  int c0 = blockIdx.x * 32, r0 = blockIdx.y * 32;
  int tx = threadIdx.x, ty = threadIdx.y;          // (32, 8)
#pragma unroll
  for (int i = 0; i < 32; i += 8)
    tile[ty + i][tx] = src[(size_t)(r0 + ty + i) * C + c0 + tx];
  __syncthreads();
#pragma unroll
  for (int i = 0; i < 32; i += 8)
    dst[(size_t)(c0 + ty + i) * R + r0 + tx] = f2bf(tile[tx][ty + i]);
}

// ---------------------------------------------------------------------------
// WMMA GEMM: A bf16 [M][K], Wt bf16 [N][K], f32 accumulate.
// Block 256 = 8 waves, tile 128x64, BK=32, double-buffered async LDS staging.
// ACT: 0 = silu, split -> (out0 fp32 + out0b bf16 | out1), 1 = softplus, 2 = +bias
// ---------------------------------------------------------------------------
template <int ACT>
__global__ __launch_bounds__(256)
void gemm_bf16_act(const __bf16* __restrict__ A, const __bf16* __restrict__ Wt,
                   const float* __restrict__ bias, float* __restrict__ out0,
                   float* __restrict__ out1, __bf16* __restrict__ out0b,
                   int M, int N, int K, int split) {
  constexpr int BM = 128, BN = 64, BK = 32, LDK = 40;  // 80B padded rows
  __shared__ __align__(16) __bf16 As[2][BM * LDK];
  __shared__ __align__(16) __bf16 Bs[2][BN * LDK];

  const int tid   = threadIdx.x;
  const int m0    = blockIdx.y * BM;
  const int n0    = blockIdx.x * BN;
  const int lane  = tid & 31;
  const int w     = tid >> 5;          // 8 waves
  const int wm    = (w & 3) * 32;      // 4x2 wave grid, 32x32 per wave
  const int wn    = (w >> 2) * 32;
  const int lrow  = lane & 15;
  const int khalf = lane >> 4;

  // Per-thread staging chunks (16B each): A = 512 chunks, B = 256 chunks.
  const int qa0_row = (tid) >> 2,        qa0_seg = (tid) & 3;
  const int qa1_row = (tid + 256) >> 2,  qa1_seg = (tid + 256) & 3;
  const int qb_row  = (tid) >> 2,        qb_seg  = (tid) & 3;

  // Running global pointers (advance 64B = BK bf16 per K-step).
  const __bf16* gA0 = A  + (size_t)(m0 + qa0_row) * K + qa0_seg * 8;
  const __bf16* gA1 = A  + (size_t)(m0 + qa1_row) * K + qa1_seg * 8;
  const __bf16* gB  = Wt + (size_t)(n0 + qb_row) * K + qb_seg * 8;
  // Precomputed LDS byte offsets for both buffers.
  unsigned lA0[2], lA1[2], lB[2];
#pragma unroll
  for (int b = 0; b < 2; ++b) {
    lA0[b] = lds_addr(&As[b][qa0_row * LDK + qa0_seg * 8]);
    lA1[b] = lds_addr(&As[b][qa1_row * LDK + qa1_seg * 8]);
    lB[b]  = lds_addr(&Bs[b][qb_row  * LDK + qb_seg  * 8]);
  }

  auto issue = [&](int buf) {
    async_copy_b128(lA0[buf], gA0);
    async_copy_b128(lA1[buf], gA1);
    async_copy_b128(lB[buf],  gB);
    gA0 += BK; gA1 += BK; gB += BK;
  };

  v8f acc[2][2] = {};
  auto compute = [&](int buf) {
    v16bf af[2], bfr[2];
#pragma unroll
    for (int t = 0; t < 2; ++t) {
      af[t]  = load_frag(&As[buf][(wm + t * 16 + lrow) * LDK], khalf);
      bfr[t] = load_frag(&Bs[buf][(wn + t * 16 + lrow) * LDK], khalf);
    }
#pragma unroll
    for (int tm = 0; tm < 2; ++tm)
#pragma unroll
      for (int tn = 0; tn < 2; ++tn)
        acc[tm][tn] = __builtin_amdgcn_wmma_f32_16x16x32_bf16(
            false, af[tm], false, bfr[tn], (short)0, acc[tm][tn], false, false);
  };

  const int nk = K / BK;
  issue(0);
  int cur = 0;
  for (int step = 0; step < nk - 1; ++step) {       // branch-free steady state
    wait_async0();          // this wave's copies into buf[cur] are done
    __syncthreads();        // ... and every other wave's too
    issue(cur ^ 1);         // overlap next-tile copy with WMMA
    compute(cur);
    __syncthreads();        // all waves done reading buf[cur] before reuse
    cur ^= 1;
  }
  wait_async0();            // peeled last tile: no further prefetch
  __syncthreads();
  compute(cur);

  // Epilogue: D layout is vgpr r -> M = khalf*8 + r, lane&15 -> N.
#pragma unroll
  for (int tm = 0; tm < 2; ++tm)
#pragma unroll
    for (int tn = 0; tn < 2; ++tn) {
      int gn = n0 + wn + tn * 16 + lrow;
      float bv = bias[gn];
#pragma unroll
      for (int r = 0; r < 8; ++r) {
        int gm = m0 + wm + tm * 16 + khalf * 8 + r;
        float v = acc[tm][tn][r] + bv;
        if (ACT == 0) {
          v = siluf_(v);
          if (gn < split) {
            out0 [(size_t)gm * split + gn] = v;
            out0b[(size_t)gm * split + gn] = f2bf(v);
          } else {
            out1[(size_t)gm * split + (gn - split)] = v;
          }
        } else if (ACT == 1) {
          out0[(size_t)gm * N + gn] = softplusf_(v);
        } else {
          out0[(size_t)gm * N + gn] = v;
        }
      }
    }
}

// K2: BC[m, 0..31] = XC[m,:] @ W_xproj + b_xproj. One wave per token row,
// lane = output column (exactly 32 columns = wave32).
__global__ __launch_bounds__(256)
void xproj_kernel(const float* __restrict__ XC, const float* __restrict__ W,
                  const float* __restrict__ bias, float* __restrict__ BC) {
  int row  = (int)((blockIdx.x * blockDim.x + threadIdx.x) >> 5);
  int lane = threadIdx.x & 31;
  if (row >= MTOK) return;
  const float* xr = XC + (size_t)row * D_INNER;
  float acc = bias[lane];
  for (int k = 0; k < D_INNER; k += 8) {
#pragma unroll
    for (int j = 0; j < 8; ++j)
      acc = fmaf(xr[k + j], W[(size_t)(k + j) * 32 + lane], acc);
  }
  BC[(size_t)row * 32 + lane] = acc;
}

// K4: selective scan. Each lane-quad owns one (b,d) channel; 4 states/lane.
// y reduction across the quad via wave32 __shfl_xor. Y may alias DT: every
// DT element is read (by its own quad, same wave) before Y is stored.
__global__ __launch_bounds__(256)
void scan_kernel(const float* __restrict__ XC, const float* DT,
                 const float* __restrict__ ZS, const float* __restrict__ BC,
                 const float* __restrict__ A_log, const float* __restrict__ Dp,
                 float* Y, __bf16* __restrict__ Yb) {
  int g    = blockIdx.x * blockDim.x + threadIdx.x;   // 0 .. 32767
  int quad = g & 3;
  int gd   = g >> 2;
  int b    = gd >> 11;            // / D_INNER
  int d    = gd & (D_INNER - 1);

  float Av[4], h[4] = {0.f, 0.f, 0.f, 0.f};
#pragma unroll
  for (int j = 0; j < 4; ++j)
    Av[j] = -__expf(A_log[(size_t)d * D_STATE + quad * 4 + j]);
  float Dv = Dp[d];

  const size_t rowbase = (size_t)b * SEQ;
  for (int t = 0; t < SEQ; ++t) {
    size_t idx   = (rowbase + t) * D_INNER + d;
    size_t bcrow = (rowbase + t) * 32 + quad * 4;
    float u   = XC[idx];
    float dtv = DT[idx];
    float du  = dtv * u;
    float y   = 0.f;
#pragma unroll
    for (int j = 0; j < 4; ++j) {
      float Bv = BC[bcrow + j];
      float Cv = BC[bcrow + 16 + j];
      float dA = __expf(dtv * Av[j]);
      h[j] = fmaf(h[j], dA, du * Bv);
      y    = fmaf(h[j], Cv, y);
    }
    y += __shfl_xor(y, 1);
    y += __shfl_xor(y, 2);        // all 4 quad lanes now hold the full sum
    if (quad == 0) {
      float yv = (y + Dv * u) * ZS[idx];
      Y[idx]  = yv;
      Yb[idx] = f2bf(yv);
    }
  }
}

extern "C" void kernel_launch(void* const* d_in, const int* in_sizes, int n_in,
                              void* d_out, int out_size, void* d_ws, size_t ws_size,
                              hipStream_t stream) {
  (void)in_sizes; (void)n_in; (void)out_size; (void)ws_size;
  const float* x       = (const float*)d_in[0];
  const float* W_in    = (const float*)d_in[1];
  const float* b_in    = (const float*)d_in[2];
  const float* W_xproj = (const float*)d_in[3];
  const float* b_xproj = (const float*)d_in[4];
  const float* W_dt    = (const float*)d_in[5];
  const float* b_dt    = (const float*)d_in[6];
  const float* W_out   = (const float*)d_in[7];
  const float* b_out   = (const float*)d_in[8];
  const float* A_log   = (const float*)d_in[9];
  const float* D_param = (const float*)d_in[10];
  float* out = (float*)d_out;

  // Workspace layout, MB-aligned offsets:
  //   XC 64 | ZS 64 | DT/Y 64 | BC 1 | Xb 16 | XCb 32 | Yb 32 | WinT 8 | WdtT 8 | WoutT 4
  char* ws = (char*)d_ws;
  const size_t MB = 1048576;
  float*  XC    = (float*)(ws);
  float*  ZS    = (float*)(ws + 64 * MB);
  float*  DTY   = (float*)(ws + 128 * MB);
  float*  BC    = (float*)(ws + 192 * MB);
  __bf16* Xb    = (__bf16*)(ws + 193 * MB);
  __bf16* XCb   = (__bf16*)(ws + 209 * MB);
  __bf16* Yb    = (__bf16*)(ws + 241 * MB);
  __bf16* WinT  = (__bf16*)(ws + 273 * MB);
  __bf16* WdtT  = (__bf16*)(ws + 281 * MB);
  __bf16* WoutT = (__bf16*)(ws + 289 * MB);

  // P0: convert x and transpose-convert weights to bf16 [N][K].
  cvt_f32_bf16<<<dim3((MTOK * D_MODEL / 4) / 256), 256, 0, stream>>>(
      x, Xb, MTOK * D_MODEL / 4);
  transpose_cvt<<<dim3((2 * D_INNER) / 32, D_MODEL / 32), dim3(32, 8), 0, stream>>>(
      W_in, WinT, D_MODEL, 2 * D_INNER);
  transpose_cvt<<<dim3(D_INNER / 32, D_INNER / 32), dim3(32, 8), 0, stream>>>(
      W_dt, WdtT, D_INNER, D_INNER);
  transpose_cvt<<<dim3(D_MODEL / 32, D_INNER / 32), dim3(32, 8), 0, stream>>>(
      W_out, WoutT, D_INNER, D_MODEL);

  // K1: x @ W_in + b_in, silu, split -> XC fp32 + XCb bf16 | ZS fp32
  gemm_bf16_act<0><<<dim3((2 * D_INNER) / 64, MTOK / 128), 256, 0, stream>>>(
      Xb, WinT, b_in, XC, ZS, XCb, MTOK, 2 * D_INNER, D_MODEL, D_INNER);
  // K2: XC @ W_xproj + b_xproj -> BC (B|C)
  xproj_kernel<<<dim3(MTOK / 8), 256, 0, stream>>>(XC, W_xproj, b_xproj, BC);
  // K3: softplus(XCb @ W_dtT + b_dt) -> DT
  gemm_bf16_act<1><<<dim3(D_INNER / 64, MTOK / 128), 256, 0, stream>>>(
      XCb, WdtT, b_dt, DTY, nullptr, nullptr, MTOK, D_INNER, D_INNER, 0);
  // K4: selective scan + gating -> Y (aliases DT buffer) + Yb bf16
  scan_kernel<<<dim3((MTOK * 4) / 256), 256, 0, stream>>>(
      XC, DTY, ZS, BC, A_log, D_param, DTY, Yb);
  // K5: Yb @ W_outT + b_out -> out
  gemm_bf16_act<2><<<dim3(D_MODEL / 64, MTOK / 128), 256, 0, stream>>>(
      Yb, WoutT, b_out, out, nullptr, nullptr, MTOK, D_MODEL, D_INNER, 0);
}
